// S4DKernel_62251255988362
// MI455X (gfx1250) — compile-verified, matrix operand order checked by reading, not run on hardware
//
#include <hip/hip_runtime.h>

typedef _Float16 f16;
typedef __attribute__((ext_vector_type(16))) _Float16 v16h;
typedef __attribute__((ext_vector_type(8)))  float    v8f;

#define D_MODEL 256
#define N2      32
#define SEQL    2048
#define NBATCH  8
#define DT      0.1f
#define NFREQ   2049        // L+1 rfft bins
#define PADF    2080        // NFREQ padded to multiple of 32 (65 chunks)
#define KEXT    (2*PADF)    // 4160: [cos rows | sin rows], 130 chunks
#define NCHUNK  (KEXT/32)   // 130
#define PI_F    3.14159265358979323846f

// ---- WMMA fragment mapping (CDNA5 ISA 7.12.2, 16-bit operands) ----
// lane: lo = lane&15, hi = lane>>4.  Element e (VGPR e>>1, half e&1) holds
// K = ((e>>3)<<4) + (hi<<3) + (e&7).  A: row M = lo.  B/C/D: col N = lo.
__device__ __forceinline__ int frag_K(int e, int hi) {
    return ((e >> 3) << 4) + (hi << 3) + (e & 7);
}

// Inverse map: (row K, col d) -> flat half-index in swizzled Bext buffer.
__device__ __forceinline__ size_t sw_index(int row, int d) {
    int c    = row >> 5;
    int K    = row & 31;
    int hi   = (K >> 3) & 1;
    int e    = ((K >> 4) << 3) | (K & 7);
    int lane = (hi << 4) | (d & 15);
    int nt   = d >> 4;
    return ((((size_t)c * 16 + nt) * 32 + lane) << 4) + e;
}

union frag_u { v16h h; uint4 q[2]; };

// ============================================================================
// Kernel 1: Cauchy sum -> swizzled Bext (f16 fragment layout).
// cos rows [0,2049): w*Re(k_f)/4096 ; sin rows base PADF: -w*Im(k_f)/4096.
// grid = PADF blocks (f), 256 threads (d).
// ============================================================================
__global__ void kf_bext_kernel(const float* __restrict__ Ar,
                               const float* __restrict__ Ai,
                               const float* __restrict__ Cr,
                               const float* __restrict__ Ci,
                               f16* __restrict__ Bsw) {
    int f = blockIdx.x;      // 0..2079
    int d = threadIdx.x;     // 0..255
    float kre = 0.f, kim = 0.f;
    if (f < NFREQ) {
        float ow_s, ow_c;
        __sincosf(-PI_F * (float)f / (float)SEQL, &ow_s, &ow_c); // e^{-i pi f/L}
        #pragma unroll 4
        for (int n = 0; n < N2; ++n) {
            int idx = d * N2 + n;
            float mag = __expf(-DT * __expf(Ar[idx]));
            float as_, ac_;
            __sincosf(DT * Ai[idx], &as_, &ac_);
            float are = mag * ac_, aim = mag * as_;      // A_disc
            float wre = ow_c * are - ow_s * aim;         // omega*A
            float wim = ow_c * aim + ow_s * are;
            float dre = 1.f - wre, dim = -wim;
            float inv = 1.f / (dre * dre + dim * dim);
            float cr = Cr[idx], ci = Ci[idx];
            kre += (cr * dre + ci * dim) * inv;
            kim += (ci * dre - cr * dim) * inv;
        }
        float w  = (f == 0 || f == SEQL) ? 1.0f : 2.0f;
        float sc = w * (1.0f / (2.0f * (float)SEQL));
        kre *= sc; kim *= sc;
    }
    Bsw[sw_index(f, d)]        = (f16)kre;
    Bsw[sw_index(PADF + f, d)] = (f16)(-kim);
}

// ============================================================================
// Kernel 2: u (B,L,D) f32 -> uT16 [d][b][s] f16 via LDS-tiled transpose.
// grid = 256 blocks: dg = blk&7 (32 channels), sr = blk>>3 (64 timesteps).
// ============================================================================
__global__ void u_to_f16_kernel(const float* __restrict__ u, f16* __restrict__ uT) {
    __shared__ __attribute__((aligned(16))) f16 tile[32][NBATCH][64]; // 32 KB
    int tid = threadIdx.x;
    int d0 = (blockIdx.x & 7) * 32;
    int s0 = (blockIdx.x >> 3) * 64;
    int ld = tid & 31, sx = tid >> 5;                   // 32 lanes x 8 s-slots
    #pragma unroll
    for (int b = 0; b < NBATCH; ++b)
        #pragma unroll
        for (int ss = 0; ss < 8; ++ss) {
            int sl = sx * 8 + ss;
            float v = u[((size_t)b * SEQL + s0 + sl) * D_MODEL + d0 + ld];
            tile[ld][b][sl] = (f16)v;
        }
    __syncthreads();
    int d = tid >> 3, b = tid & 7;                      // one 128B row per thread
    const uint4* src = (const uint4*)&tile[d][b][0];
    uint4* dst = (uint4*)(uT + ((size_t)(d0 + d) * NBATCH + b) * SEQL + s0);
    #pragma unroll
    for (int i = 0; i < 8; ++i) dst[i] = src[i];
}

// ============================================================================
// Kernel 3: k(2048 x 256) = A_basis(2048 x 4160) @ Bext(4160 x 256) via WMMA.
// A from 4096-entry LDS cos table; sin half = cos table with +3072 phase.
// grid = 128 blocks (t-tile), 256 threads = 8 waves, 2 n-tiles per wave.
// ============================================================================
__global__ void kgen_kernel(const f16* __restrict__ Bsw, f16* __restrict__ k16) {
    __shared__ __attribute__((aligned(16))) f16 costab[4096];
    int tid = threadIdx.x;
    for (int m = tid; m < 4096; m += 256)
        costab[m] = (f16)__cosf(PI_F * (float)m / (float)SEQL);
    __syncthreads();

    int t0   = blockIdx.x * 16;
    int wave = tid >> 5, lane = tid & 31;
    int lo = lane & 15, hi = lane >> 4;
    int nt0 = wave * 2;
    int t   = t0 + lo;                 // A row (M) for this lane
    int t32 = t << 5;

    int tKe[16];                       // per-element K * t (hoisted multiplies)
    #pragma unroll
    for (int e = 0; e < 16; ++e) tKe[e] = frag_K(e, hi) * t;

    v8f acc0 = {}, acc1 = {};

    auto chunk = [&](int c, int ft) {
        v16h a;
        #pragma unroll
        for (int e = 0; e < 16; ++e)
            a[e] = costab[(ft + tKe[e]) & 4095];
        frag_u b0, b1;
        const uint4* p0 = (const uint4*)(Bsw + ((((size_t)c * 16 + nt0)     * 32 + lane) << 4));
        const uint4* p1 = (const uint4*)(Bsw + ((((size_t)c * 16 + nt0 + 1) * 32 + lane) << 4));
        b0.q[0] = p0[0]; b0.q[1] = p0[1];
        b1.q[0] = p1[0]; b1.q[1] = p1[1];
        acc0 = __builtin_amdgcn_wmma_f32_16x16x32_f16(false, a, false, b0.h, (short)0, acc0, false, false);
        acc1 = __builtin_amdgcn_wmma_f32_16x16x32_f16(false, a, false, b1.h, (short)0, acc1, false, false);
    };

    int ft = 0;                                   // cos half: idx = (f*t)&4095
    for (int c = 0; c < NCHUNK / 2; ++c, ft += t32) chunk(c, ft);
    ft = 3072;                                    // sin half: cos(x+3pi/2)=sin x
    for (int c = NCHUNK / 2; c < NCHUNK; ++c, ft += t32) chunk(c, ft);

    #pragma unroll
    for (int r = 0; r < 8; ++r) {
        int tt = t0 + r + hi * 8;
        k16[(size_t)(nt0 * 16 + lo) * SEQL + tt]       = (f16)acc0[r];
        k16[(size_t)((nt0 + 1) * 16 + lo) * SEQL + tt] = (f16)acc1[r];
    }
}

// ============================================================================
// Kernel 4: depthwise conv, per-channel Toeplitz WMMA.
// y[b,t,d] = sum_s kpad[s - t + 2047] * u[b,s,d] + D[d]
// u slice staged LDS via global_load_async_to_lds_b128 (ASYNCcnt path).
// ============================================================================
__global__ void conv_kernel(const f16* __restrict__ k16,
                            const f16* __restrict__ uT,
                            const float* __restrict__ Dbias,
                            float* __restrict__ out) {
    __shared__ __attribute__((aligned(16))) f16 kpad[4096];          // 8 KB
    __shared__ __attribute__((aligned(16))) f16 uls[NBATCH * SEQL];  // 32 KB [b][s]

    int d = blockIdx.x, tid = threadIdx.x;

    // async-stage the contiguous 32KB u slice: LDS offset = base + 16B*i
    {
        unsigned long long gsrc = (unsigned long long)(uT + (size_t)d * NBATCH * SEQL);
        unsigned lds_base = (unsigned)(unsigned long long)(&uls[0]);
        for (int i = tid; i < NBATCH * SEQL / 8; i += 256) {
            unsigned ldsoff = lds_base + (unsigned)i * 16u;
            unsigned goff   = (unsigned)i * 16u;
            asm volatile("global_load_async_to_lds_b128 %0, %1, %2"
                         :: "v"(ldsoff), "v"(goff), "s"(gsrc) : "memory");
        }
    }
    // kpad[q] = k[d][q-1024] in range, else 0
    for (int i = tid; i < 4096; i += 256) {
        int tt = i - 1024;
        kpad[i] = (tt >= 0 && tt < SEQL) ? k16[(size_t)d * SEQL + tt] : (f16)0.0f;
    }
    asm volatile("s_wait_asynccnt 0x0" ::: "memory");
    __syncthreads();

    float bias = Dbias[d];
    int wave = tid >> 5, lane = tid & 31;
    int lo = lane & 15, hi = lane >> 4;

    int aKe[16];                        // per-element K (A gather offsets)
    #pragma unroll
    for (int e = 0; e < 16; ++e) aKe[e] = frag_K(e, hi);

    const f16* urow = &uls[(size_t)(lo & 7) * SEQL + (hi << 3)]; // lanes lo>=8: finite garbage, cols unused

    for (int tt = wave; tt < SEQL / 16; tt += 8) {
        int t0 = tt * 16;
        v8f acc = {};
        int smin = t0 - 1023;
        int c_lo = (smin > 0) ? (smin >> 5) : 0;
        int c_hi = ((t0 + 15 + 1024) >> 5) + 1;
        if (c_hi > SEQL / 32) c_hi = SEQL / 32;

        int base = 2047 - t0 - lo;      // kpad idx = base + s0 + K  (A row M = lo)
        for (int c = c_lo; c < c_hi; ++c) {
            int s0 = c << 5;
            v16h a;
            #pragma unroll
            for (int e = 0; e < 16; ++e)
                a[e] = kpad[base + s0 + aKe[e]];
            frag_u b;
            const uint4* p = (const uint4*)(urow + s0);
            b.q[0] = p[0];              // K = hi*8 + 0..7
            b.q[1] = p[2];              // K = 16 + hi*8 + 0..7
            acc = __builtin_amdgcn_wmma_f32_16x16x32_f16(false, a, false, b.h, (short)0, acc, false, false);
        }

        if (lo < NBATCH) {              // D layout: N = lo (batch), M = r + 8*hi
            #pragma unroll
            for (int r = 0; r < 8; ++r) {
                int tglob = t0 + r + hi * 8;
                out[((size_t)lo * SEQL + tglob) * D_MODEL + d] = acc[r] + bias;
            }
        }
    }
}

// ============================================================================
extern "C" void kernel_launch(void* const* d_in, const int* in_sizes, int n_in,
                              void* d_out, int out_size, void* d_ws, size_t ws_size,
                              hipStream_t stream) {
    (void)in_sizes; (void)n_in; (void)out_size; (void)ws_size;
    const float* u  = (const float*)d_in[0];
    const float* Ar = (const float*)d_in[1];
    const float* Ai = (const float*)d_in[2];
    const float* Cr = (const float*)d_in[3];
    const float* Ci = (const float*)d_in[4];
    const float* Dd = (const float*)d_in[5];
    float* out = (float*)d_out;

    // workspace layout (f16 elements)
    f16* Bsw = (f16*)d_ws;                                   // 130*16*32*16 = 1,064,960
    f16* k16 = Bsw + (size_t)NCHUNK * 16 * 32 * 16;          // 256*2048
    f16* uT  = k16 + (size_t)D_MODEL * SEQL;                 // 256*8*2048 (16B aligned)

    kf_bext_kernel<<<PADF, 256, 0, stream>>>(Ar, Ai, Cr, Ci, Bsw);
    u_to_f16_kernel<<<256, 256, 0, stream>>>(u, uT);
    kgen_kernel<<<SEQL / 16, 256, 0, stream>>>(Bsw, k16);
    conv_kernel<<<D_MODEL, 256, 0, stream>>>(k16, uT, Dd, out);
}
